// ResidualVectorQuantizer_90709709291573
// MI455X (gfx1250) — matrix operand order repaired and checked by reading, not compiled, not run on hardware
//
#include <hip/hip_runtime.h>
#include <hip/hip_bf16.h>

// ---------------------------------------------------------------------------
// Residual VQ forward for MI455X (gfx1250, wave32, WMMA).
// Heavy op: dist GEMM residual[16384x256] x cb^T[256x1024] per stage (x8).
// Uses V_WMMA_F32_16X16X32_BF16 with a bf16 hi/lo split (3 products) to get
// ~fp32 accuracy at bf16-matrix throughput.
// ---------------------------------------------------------------------------

typedef __attribute__((ext_vector_type(16))) __bf16 v16bf;
typedef __attribute__((ext_vector_type(8)))  __bf16 v8bf;
typedef __attribute__((ext_vector_type(8)))  float  v8f;

#define DEV __device__ __forceinline__

constexpr int Bn   = 8;
constexpr int C    = 256;
constexpr int T    = 2048;
constexpr int NQ   = 8;
constexpr int BINS = 1024;
constexpr int ROWS = Bn * T;          // 16384 (B*T)
constexpr int NBLK = ROWS / 64;       // 256 blocks of 64 rows

// 64 x 256 f32 tile in LDS (exactly 64KB). XOR/rotate swizzle keeps both
// row-contiguous and column-strided accesses bank-conflict free.
DEV int sw(int r, int c) { return r * C + ((c + r) & (C - 1)); }
constexpr size_t TILE_LDS_BYTES = (size_t)64 * C * sizeof(float);

DEV unsigned short f32_to_bf16(float f) {
  unsigned u = __float_as_uint(f);
  u += 0x7FFFu + ((u >> 16) & 1u);            // round-to-nearest-even
  return (unsigned short)(u >> 16);
}
DEV float bf16_to_f32(unsigned short h) { return __uint_as_float(((unsigned)h) << 16); }

DEV float wave_sum(float v) {
#pragma unroll
  for (int m = 16; m > 0; m >>= 1) v += __shfl_xor(v, m, 32);
  return v;
}

DEV v8bf ld8(const unsigned short* p) { return *reinterpret_cast<const v8bf*>(p); }
DEV v16bf make16(v8bf a, v8bf b) {
  v16bf r;
#pragma unroll
  for (int i = 0; i < 8; ++i) { r[i] = a[i]; r[i + 8] = b[i]; }
  return r;
}

// ---------------------------------------------------------------------------
// k_init: x[B,C,T] -> residual[B*T, C] (f32 + bf16 hi/lo), r2[row] = ||r||^2.
// ---------------------------------------------------------------------------
__global__ void k_init(const float* __restrict__ x, float* __restrict__ residual,
                       unsigned short* __restrict__ res_hi,
                       unsigned short* __restrict__ res_lo,
                       float* __restrict__ r2) {
  extern __shared__ float s[];
  const int tid = threadIdx.x;
  const int base = blockIdx.x * 64;              // rows base..base+63, same b
  const int b = base / T, tBase = base % T;
  const int tl = tid & 63, cg = tid >> 6;
#pragma unroll 4
  for (int cc = 0; cc < 64; ++cc) {              // coalesced over t
    const int c = cc * 4 + cg;
    s[sw(tl, c)] = x[(size_t)b * C * T + (size_t)c * T + tBase + tl];
  }
  __syncthreads();
  const int wave = tid >> 5, lane = tid & 31;
  for (int j = 0; j < 8; ++j) {
    const int r = wave * 8 + j, row = base + r;
    float acc = 0.f;
#pragma unroll
    for (int jj = 0; jj < 8; ++jj) {
      const int c = lane + 32 * jj;              // coalesced over c
      const float v = s[sw(r, c)];
      const size_t o = (size_t)row * C + c;
      residual[o] = v;
      const unsigned short h = f32_to_bf16(v);
      res_hi[o] = h;
      res_lo[o] = f32_to_bf16(v - bf16_to_f32(h));
      acc += v * v;
    }
    acc = wave_sum(acc);
    if (lane == 0) r2[row] = acc;
  }
}

// ---------------------------------------------------------------------------
// k_cb: codebooks[NQ*BINS, C] -> bf16 hi/lo + e2[bin] = ||e||^2. One block/bin.
// ---------------------------------------------------------------------------
__global__ void k_cb(const float* __restrict__ cbs, unsigned short* __restrict__ cb_hi,
                     unsigned short* __restrict__ cb_lo, float* __restrict__ e2) {
  __shared__ float sred[8];
  const int g = blockIdx.x;                      // stage*BINS + bin
  const int tid = threadIdx.x;                   // = c
  const size_t o = (size_t)g * C + tid;
  const float v = cbs[o];
  const unsigned short h = f32_to_bf16(v);
  cb_hi[o] = h;
  cb_lo[o] = f32_to_bf16(v - bf16_to_f32(h));
  const float ssum = wave_sum(v * v);
  if ((tid & 31) == 0) sred[tid >> 5] = ssum;
  __syncthreads();
  if (tid == 0) {
    float t = 0.f;
    for (int i = 0; i < 8; ++i) t += sred[i];
    e2[g] = t;
  }
}

// ---------------------------------------------------------------------------
// k_dist: fused dist GEMM + dists write + argmin.
// Block = 64 rows x all 1024 bins, K=256. 8 waves: wave w -> row tile w&3,
// col half w>>2 (64 cols), 4 col tiles of 16 per wave per 128-col block.
// dot accumulated as hi*hi + hi*lo + lo*hi (3 bf16 WMMAs per k-step).
// ---------------------------------------------------------------------------
__global__ void k_dist(const unsigned short* __restrict__ res_hi,
                       const unsigned short* __restrict__ res_lo,
                       const unsigned short* __restrict__ cb_hi,
                       const unsigned short* __restrict__ cb_lo,
                       const float* __restrict__ r2,
                       const float* __restrict__ e2,
                       float* __restrict__ dists, int* __restrict__ idx_out,
                       int stage) {
  __shared__ unsigned long long sMin[64];
  const int tid = threadIdx.x;
  if (tid < 64) sMin[tid] = ~0ull;
  __syncthreads();

  const int wave = tid >> 5, lane = tid & 31;
  const int rw = wave & 3, cw = wave >> 2;
  const int half = lane >> 4, ln = lane & 15;
  const int rowBase = blockIdx.x * 64;

  // A-frag source: row = rowBase + rw*16 + ln (A layout: lane = M, runs of 8 K)
  const unsigned short* aH = res_hi + (size_t)(rowBase + rw * 16 + ln) * C;
  const unsigned short* aL = res_lo + (size_t)(rowBase + rw * 16 + ln) * C;
  const unsigned short* bH = cb_hi + (size_t)stage * BINS * C;
  const unsigned short* bL = cb_lo + (size_t)stage * BINS * C;
  const float* e2s = e2 + stage * BINS;

  float r2v[8];
#pragma unroll
  for (int v = 0; v < 8; ++v) r2v[v] = r2[rowBase + rw * 16 + half * 8 + v];

  unsigned long long mink[8];
#pragma unroll
  for (int v = 0; v < 8; ++v) mink[v] = ~0ull;

  const size_t dbase = (size_t)stage * ROWS * BINS;

  for (int cbk = 0; cbk < 8; ++cbk) {            // 128-col blocks
    const int colBase = cbk * 128 + cw * 64;
    v8f acc[4] = {};
    for (int ks = 0; ks < 8; ++ks) {             // K steps of 32
      const int k0 = ks * 32;
      // A 16x32 bf16 layout: lane half selects K runs {h*8..h*8+7, 16+h*8..}
      const v16bf ah = make16(ld8(aH + k0 + half * 8), ld8(aH + k0 + 16 + half * 8));
      const v16bf al = make16(ld8(aL + k0 + half * 8), ld8(aL + k0 + 16 + half * 8));
#pragma unroll
      for (int ct = 0; ct < 4; ++ct) {
        const int n = colBase + ct * 16 + ln;    // B: lane = N, K runs {h*8..h*8+7, 16+h*8..}
        const unsigned short* pb = bH + (size_t)n * C + k0 + half * 8;
        const unsigned short* pl = bL + (size_t)n * C + k0 + half * 8;
        const v16bf bh = make16(ld8(pb), ld8(pb + 16));
        const v16bf bl = make16(ld8(pl), ld8(pl + 16));
        acc[ct] = __builtin_amdgcn_wmma_f32_16x16x32_bf16(false, ah, false, bh,
                                                          (short)0, acc[ct], false, false);
        acc[ct] = __builtin_amdgcn_wmma_f32_16x16x32_bf16(false, ah, false, bl,
                                                          (short)0, acc[ct], false, false);
        acc[ct] = __builtin_amdgcn_wmma_f32_16x16x32_bf16(false, al, false, bh,
                                                          (short)0, acc[ct], false, false);
      }
    }
    // finalize: dist = ||r||^2 + ||e||^2 - 2*dot; write + running argmin
#pragma unroll
    for (int ct = 0; ct < 4; ++ct) {
      const int n = colBase + ct * 16 + ln;
      const float e2n = e2s[n];
#pragma unroll
      for (int v = 0; v < 8; ++v) {
        const int row = rowBase + rw * 16 + half * 8 + v;  // D: M = v + 8*half
        const float d = r2v[v] + e2n - 2.0f * acc[ct][v];
        dists[dbase + (size_t)row * BINS + n] = d;
        const unsigned u = __float_as_uint(d);
        const unsigned key32 = (u & 0x80000000u) ? ~u : (u | 0x80000000u);
        const unsigned long long key = ((unsigned long long)key32 << 32) | (unsigned)n;
        mink[v] = key < mink[v] ? key : mink[v];
      }
    }
  }
#pragma unroll
  for (int v = 0; v < 8; ++v)
    atomicMin(&sMin[rw * 16 + half * 8 + v], mink[v]);   // ds_min_u64
  __syncthreads();
  if (tid < 64) idx_out[rowBase + tid] = (int)(sMin[tid] & 0xFFFFFFFFu);
}

// ---------------------------------------------------------------------------
// k_update: quant = cb[idx]; residual -= quant (+ new hi/lo + r2);
// quantized += quant; sub_quants/codes written in [B,C,T] via swizzled LDS.
// ---------------------------------------------------------------------------
__global__ void k_update(float* __restrict__ residual,
                         unsigned short* __restrict__ res_hi,
                         unsigned short* __restrict__ res_lo,
                         const float* __restrict__ cbs,
                         const int* __restrict__ idxArr,
                         float* __restrict__ out_q,
                         float* __restrict__ out_subq,
                         float* __restrict__ out_codes,
                         float* __restrict__ r2,
                         float* __restrict__ lossPartial,
                         int stage) {
  extern __shared__ float s[];
  __shared__ float sl[8];
  const int tid = threadIdx.x;
  const int base = blockIdx.x * 64;
  const int b = base / T, tBase = base % T;
  const int wave = tid >> 5, lane = tid & 31;
  const float* cbS = cbs + (size_t)stage * BINS * C;
  float lsum = 0.f;
  for (int j = 0; j < 8; ++j) {                  // wave handles 8 rows
    const int r = wave * 8 + j, row = base + r;
    const float* q = cbS + (size_t)idxArr[row] * C;
    float racc = 0.f;
#pragma unroll
    for (int jj = 0; jj < 8; ++jj) {
      const int c = lane + 32 * jj;
      const size_t o = (size_t)row * C + c;
      const float rv = residual[o];
      const float qv = q[c];
      const float nr = rv - qv;
      residual[o] = nr;
      const unsigned short h = f32_to_bf16(nr);
      res_hi[o] = h;
      res_lo[o] = f32_to_bf16(nr - bf16_to_f32(h));
      s[sw(r, c)] = qv;
      lsum += (qv - rv) * (qv - rv);
      racc += nr * nr;
    }
    racc = wave_sum(racc);
    if (lane == 0) r2[row] = racc;
  }
  lsum = wave_sum(lsum);
  if (lane == 0) sl[wave] = lsum;
  if (tid < 64) out_codes[(size_t)stage * ROWS + base + tid] = (float)idxArr[base + tid];
  __syncthreads();
  if (tid == 0) {                                // deterministic loss partial
    float t = 0.f;
    for (int i = 0; i < 8; ++i) t += sl[i];
    lossPartial[stage * NBLK + blockIdx.x] = t;
  }
  // write quantized / sub_quants in [B,C,T] (t contiguous per thread group)
  const int tl = tid & 63, cg = tid >> 6;
  const size_t sqb = (size_t)stage * Bn * C * T;
#pragma unroll 4
  for (int cc = 0; cc < 64; ++cc) {
    const int c = cc * 4 + cg;
    const float qv = s[sw(tl, c)];
    const size_t o = (size_t)b * C * T + (size_t)c * T + tBase + tl;
    out_subq[sqb + o] = qv;
    if (stage == 0) out_q[o] = qv;
    else            out_q[o] += qv;
  }
}

// ---------------------------------------------------------------------------
// k_final: penalty = sum(loss_i)/nnz, bw = NQ*log2(BINS)*16000/320 = 4000.
// ---------------------------------------------------------------------------
__global__ void k_final(const float* __restrict__ lossPartial,
                        float* __restrict__ out_bw, float* __restrict__ out_pen) {
  float sum = 0.f;
  int nnz = 0;
  for (int i = 0; i < NQ; ++i) {
    float l = 0.f;
    for (int p = 0; p < NBLK; ++p) l += lossPartial[i * NBLK + p];
    l *= 0.1f / ((float)ROWS * (float)C);        // COMMITMENT_WEIGHT * mean
    sum += l;
    if (l != 0.f) nnz++;
  }
  *out_pen = sum / (float)nnz;
  *out_bw = 4000.0f;
}

// ---------------------------------------------------------------------------
extern "C" void kernel_launch(void* const* d_in, const int* in_sizes, int n_in,
                              void* d_out, int out_size, void* d_ws, size_t ws_size,
                              hipStream_t stream) {
  const float* x   = (const float*)d_in[0];
  const float* cbs = (const float*)d_in[2];      // d_in[1] = sample_rate (const)
  float* out = (float*)d_out;

  // Output layout (flat, reference return order)
  float* out_q     = out;                                  // [B,C,T]
  float* out_codes = out_q + (size_t)Bn * C * T;           // [NQ,B,T] as float
  float* out_bw    = out_codes + (size_t)NQ * Bn * T;      // scalar
  float* out_pen   = out_bw + 1;                           // scalar
  float* out_subq  = out_pen + 1;                          // [NQ,B,C,T]
  float* out_dists = out_subq + (size_t)NQ * Bn * C * T;   // [NQ,B,T,BINS]

  // Workspace carve-out (~42 MB)
  char* ws = (char*)d_ws;
  size_t off = 0;
  auto take = [&](size_t bytes) -> void* {
    void* p = ws + off;
    off += (bytes + 255) & ~(size_t)255;
    return p;
  };
  float*          residual = (float*)take((size_t)ROWS * C * 4);
  unsigned short* res_hi   = (unsigned short*)take((size_t)ROWS * C * 2);
  unsigned short* res_lo   = (unsigned short*)take((size_t)ROWS * C * 2);
  unsigned short* cb_hi    = (unsigned short*)take((size_t)NQ * BINS * C * 2);
  unsigned short* cb_lo    = (unsigned short*)take((size_t)NQ * BINS * C * 2);
  float*          e2       = (float*)take((size_t)NQ * BINS * 4);
  float*          r2       = (float*)take((size_t)ROWS * 4);
  int*            idxArr   = (int*)take((size_t)ROWS * 4);
  float*          lossPart = (float*)take((size_t)NQ * NBLK * 4);

  k_init<<<NBLK, 256, TILE_LDS_BYTES, stream>>>(x, residual, res_hi, res_lo, r2);
  k_cb<<<NQ * BINS, 256, 0, stream>>>(cbs, cb_hi, cb_lo, e2);
  for (int i = 0; i < NQ; ++i) {
    k_dist<<<NBLK, 256, 0, stream>>>(res_hi, res_lo, cb_hi, cb_lo, r2, e2,
                                     out_dists, idxArr, i);
    k_update<<<NBLK, 256, TILE_LDS_BYTES, stream>>>(residual, res_hi, res_lo, cbs,
                                                    idxArr, out_q, out_subq, out_codes,
                                                    r2, lossPart, i);
  }
  k_final<<<1, 1, 0, stream>>>(lossPart, out_bw, out_pen);

  (void)in_sizes; (void)n_in; (void)out_size; (void)ws_size;
}
